// SFI_326417514719
// MI455X (gfx1250) — compile-verified, matrix operand order checked by reading, not run on hardware
//
#include <hip/hip_runtime.h>
#include <math.h>

// ---------------------------------------------------------------------------
// gfx1250 WMMA bf16 path: V_WMMA_F32_16X16X32_BF16
//   A: 16x32 bf16 (16 bf16/lane = 8 VGPRs), B: 32x16 bf16, C/D: 8xf32.
// bf16 kept as unsigned short; bit-cast into the builtin operand type.
// ---------------------------------------------------------------------------
typedef __attribute__((ext_vector_type(16))) __bf16          bf16x16;
typedef __attribute__((ext_vector_type(8)))  unsigned short  us8;
typedef __attribute__((ext_vector_type(4)))  unsigned short  us4;
typedef __attribute__((ext_vector_type(4)))  float           v4f;
typedef __attribute__((ext_vector_type(8)))  float           v8f;

union FragU { us8 h[2]; bf16x16 b; };

__device__ __forceinline__ unsigned short f2bf(float x) {
    unsigned int u = __float_as_uint(x);
    u += 0x7FFFu + ((u >> 16) & 1u);   // round-to-nearest-even
    return (unsigned short)(u >> 16);
}

// CDNA5 async copy global -> LDS (16 bytes), tracked by ASYNCcnt.
// lds_addr: LDS byte address (low 32 bits of flat shared pointer, ISA 10.2)
// voff:     per-lane 32-bit unsigned byte offset;  sbase: uniform 64-bit base
__device__ __forceinline__ void async_ld16(unsigned lds_addr, unsigned voff,
                                           const void* sbase) {
    asm volatile("global_load_async_to_lds_b128 %0, %1, %2"
                 :: "v"(lds_addr), "v"(voff), "s"(sbase) : "memory");
}
__device__ __forceinline__ void wait_async0() {
    asm volatile("s_wait_asynccnt 0x0" ::: "memory");
}

// ---------------------------------------------------------------------------
// f32 -> bf16 conversion, 4 elements per thread (n must be multiple of 4)
// ---------------------------------------------------------------------------
__global__ void f32_to_bf16_v4_kernel(const v4f* __restrict__ src,
                                      us4* __restrict__ dst, long n4) {
    long i = (long)blockIdx.x * blockDim.x + threadIdx.x;
    if (i < n4) {
        v4f v = src[i];
        us4 o;
        o.x = f2bf(v.x); o.y = f2bf(v.y); o.z = f2bf(v.z); o.w = f2bf(v.w);
        dst[i] = o;
    }
}

// ---------------------------------------------------------------------------
// Row softmax over D=512 (one block of 256 threads per row), bf16 output
// ---------------------------------------------------------------------------
__global__ void softmax512_kernel(const float* __restrict__ scores,
                                  unsigned short* __restrict__ attn) {
    __shared__ float red[256];
    const int D = 512;
    long row = blockIdx.x;
    const float* s = scores + row * (long)D;
    int tid = threadIdx.x;
    float v0 = s[tid], v1 = s[tid + 256];
    red[tid] = fmaxf(v0, v1);
    __syncthreads();
    for (int off = 128; off > 0; off >>= 1) {
        if (tid < off) red[tid] = fmaxf(red[tid], red[tid + off]);
        __syncthreads();
    }
    float rmax = red[0];
    __syncthreads();
    float e0 = __expf(v0 - rmax), e1 = __expf(v1 - rmax);
    red[tid] = e0 + e1;
    __syncthreads();
    for (int off = 128; off > 0; off >>= 1) {
        if (tid < off) red[tid] += red[tid + off];
        __syncthreads();
    }
    float inv = 1.0f / red[0];
    unsigned short* o = attn + row * (long)D;
    o[tid]       = f2bf(e0 * inv);
    o[tid + 256] = f2bf(e1 * inv);
}

// ---------------------------------------------------------------------------
// Batched bf16 WMMA GEMM, D = A(MxK)*B(KxN) + epilogue, tile 128x128x32,
// double-buffered LDS fed by global_load_async_to_lds_b128.
//  BMODE 0: B supplied as B^T, row-major (N x K), ld = ldb   (async copy)
//  BMODE 1: B supplied row-major (K x N), ld = ldb           (sync scatter)
//  EPI  0: out_bf16[gm][gn] = acc + biasRow[gm]
//       1: out_f32 [gm][gn] = acc * scale
//       2: out_bf16[gn][gm] = acc + resid[gm][gn]      (transposed store)
//       3: out_bf16[gm][gn] = gelu(acc + biasCol[gn])  (exact erf gelu)
//       4: out_f32 [gm][gn] = (acc + biasRow[gm] + resid[gm][gn]) * mask[gn]
// All dims are multiples of the tile here (512 / 2048).
// ---------------------------------------------------------------------------
template <int BMODE, int EPI>
__global__ __launch_bounds__(256)
void gemm_bf16_wmma_kernel(const unsigned short* __restrict__ A,
                           const unsigned short* __restrict__ B,
                           void* __restrict__ Out,
                           const float* __restrict__ biasRow,
                           const float* __restrict__ biasCol,
                           const float* __restrict__ resid,
                           const float* __restrict__ maskp,
                           float scale, int K,
                           int lda, int ldb, int ldo, int ldr,
                           long sA, long sB, long sO, long sR, int Tmask) {
    constexpr int BM = 128, BN = 128, BK = 32, LROW = 40;  // LROW: +8 pad
    constexpr unsigned LDSBUF = (unsigned)(BM * LROW * 2); // bytes per buffer
    __shared__ __align__(16) unsigned short As[2][BM][LROW];
    __shared__ __align__(16) unsigned short Bs[2][BN][LROW];

    const int nb  = blockIdx.z;
    const int m0  = blockIdx.y * BM;
    const int n0  = blockIdx.x * BN;
    const int tid  = threadIdx.x;
    const int lane = tid & 31;
    const int wid  = tid >> 5;
    const int wm = (wid & 3) * 32;   // 4 waves along M
    const int wn = (wid >> 2) * 64;  // 2 waves along N

    const unsigned short* Ab = A + (size_t)nb * sA;
    const unsigned short* Bb = B + (size_t)nb * sB;

    // async staging geometry: thread -> rows tq, tq+64 ; 16-byte column chunk
    const int tq = tid >> 2;              // 0..63
    const int kq = (tid & 3) * 8;         // element offset inside BK
    const unsigned offA0 = (unsigned)(((size_t)(m0 + tq) * lda + kq) * 2);
    const unsigned offA1 = offA0 + (unsigned)(64u * (unsigned)lda * 2u);
    const unsigned ldsA0 = (unsigned)(size_t)&As[0][tq][kq];
    const unsigned ldsA1 = (unsigned)(size_t)&As[0][tq + 64][kq];
    unsigned offB0 = 0, offB1 = 0, ldsB0 = 0, ldsB1 = 0;
    if (BMODE == 0) {
        offB0 = (unsigned)(((size_t)(n0 + tq) * ldb + kq) * 2);
        offB1 = offB0 + (unsigned)(64u * (unsigned)ldb * 2u);
        ldsB0 = (unsigned)(size_t)&Bs[0][tq][kq];
        ldsB1 = (unsigned)(size_t)&Bs[0][tq + 64][kq];
    }

    auto stage_async = [&](unsigned bufofs, int kk) {
        const unsigned koff = (unsigned)kk * 2u;
        async_ld16(ldsA0 + bufofs, offA0 + koff, Ab);
        async_ld16(ldsA1 + bufofs, offA1 + koff, Ab);
        if (BMODE == 0) {
            async_ld16(ldsB0 + bufofs, offB0 + koff, Bb);
            async_ld16(ldsB1 + bufofs, offB1 + koff, Bb);
        }
    };
    auto stage_b_sync = [&](int bufw, int kk) {
        // B row-major (K x N): vector read along N, scatter into N-major LDS
        const int kr   = tid >> 4;         // 0..15
        const int nofs = (tid & 15) * 8;
#pragma unroll
        for (int j = 0; j < 2; ++j) {
            int k = kr + 16 * j;
            us8 v = *(const us8*)(Bb + (size_t)(kk + k) * ldb + (n0 + nofs));
#pragma unroll
            for (int e = 0; e < 8; ++e) Bs[bufw][nofs + e][k] = v[e];
        }
    };

    // ---- prologue: tile 0 into buffer 0 ----
    stage_async(0u, 0);
    if (BMODE == 1) stage_b_sync(0, 0);
    wait_async0();
    __syncthreads();

    v8f acc[2][4];
    const v8f vzero = {0.f, 0.f, 0.f, 0.f, 0.f, 0.f, 0.f, 0.f};
#pragma unroll
    for (int mi = 0; mi < 2; ++mi)
#pragma unroll
        for (int ni = 0; ni < 4; ++ni) acc[mi][ni] = vzero;

    const int mrow = lane & 15;         // M (for A) / N (for B) index
    const int kb   = (lane >> 4) * 8;   // K base per half-wave
    int buf = 0;

    for (int k0 = 0; k0 < K; k0 += BK) {
        const int nxt = buf ^ 1;
        if (k0 + BK < K) {                       // prefetch next tile
            stage_async((unsigned)nxt * LDSBUF, k0 + BK);
            if (BMODE == 1) stage_b_sync(nxt, k0 + BK);
        }
        // fragments per the ISA 16-bit A/B VGPR layout
        FragU af[2], bfr[4];
#pragma unroll
        for (int mi = 0; mi < 2; ++mi) {
            const unsigned short* p = &As[buf][wm + mi * 16 + mrow][kb];
            af[mi].h[0] = *(const us8*)p;        // K = kb .. kb+7
            af[mi].h[1] = *(const us8*)(p + 16); // K = kb+16 .. kb+23
        }
#pragma unroll
        for (int ni = 0; ni < 4; ++ni) {
            const unsigned short* p = &Bs[buf][wn + ni * 16 + mrow][kb];
            bfr[ni].h[0] = *(const us8*)p;
            bfr[ni].h[1] = *(const us8*)(p + 16);
        }
#pragma unroll
        for (int mi = 0; mi < 2; ++mi)
#pragma unroll
            for (int ni = 0; ni < 4; ++ni)
                acc[mi][ni] = __builtin_amdgcn_wmma_f32_16x16x32_bf16(
                    false, af[mi].b, false, bfr[ni].b,
                    (short)0, acc[mi][ni], false, false);

        wait_async0();      // next tile resident in LDS
        __syncthreads();    // everyone done reading current buffer
        buf = nxt;
    }

    // ---- epilogue; C/D layout: VGPR r -> M = r + 8*(lane/16), N = lane%16 --
    const int mofs = 8 * (lane >> 4);
    const int nofs = lane & 15;
#pragma unroll
    for (int mi = 0; mi < 2; ++mi) {
#pragma unroll
        for (int ni = 0; ni < 4; ++ni) {
#pragma unroll
            for (int r = 0; r < 8; ++r) {
                int gm = m0 + wm + mi * 16 + r + mofs;
                int gn = n0 + wn + ni * 16 + nofs;
                float v = acc[mi][ni][r];
                if (EPI == 0) {
                    v += biasRow[gm];
                    ((unsigned short*)Out)[(size_t)nb * sO + (size_t)gm * ldo + gn] = f2bf(v);
                } else if (EPI == 1) {
                    ((float*)Out)[(size_t)nb * sO + (size_t)gm * ldo + gn] = v * scale;
                } else if (EPI == 2) {
                    v += resid[(size_t)nb * sR + (size_t)gm * ldr + gn];
                    ((unsigned short*)Out)[(size_t)nb * sO + (size_t)gn * ldo + gm] = f2bf(v);
                } else if (EPI == 3) {
                    v += biasCol[gn];
                    float g = 0.5f * v * (1.0f + erff(v * 0.70710678118654752f));
                    ((unsigned short*)Out)[(size_t)nb * sO + (size_t)gm * ldo + gn] = f2bf(g);
                } else { // EPI == 4
                    v += biasRow[gm] + resid[(size_t)nb * sR + (size_t)gm * ldr + gn];
                    v *= maskp[(size_t)nb * Tmask + gn];
                    ((float*)Out)[(size_t)nb * sO + (size_t)gm * ldo + gn] = v;
                }
            }
        }
    }
}

// ---------------------------------------------------------------------------
// Host-side orchestration
// ---------------------------------------------------------------------------
extern "C" void kernel_launch(void* const* d_in, const int* in_sizes, int n_in,
                              void* d_out, int out_size, void* d_ws, size_t ws_size,
                              hipStream_t stream) {
    (void)in_sizes; (void)n_in; (void)out_size; (void)ws_size;
    const int N = 16, T = 2048, C = 512, F = 512;

    const float* feature_s = (const float*)d_in[0];  // (N,T,C)
    const float* feature_t = (const float*)d_in[1];  // (N,F,T)
    const float* maskp     = (const float*)d_in[2];  // (N,1,T)
    const float* conv_w    = (const float*)d_in[3];  // (F,C)
    const float* conv_b    = (const float*)d_in[4];  // (F)
    const float* w1        = (const float*)d_in[5];  // (F,F)
    const float* b1        = (const float*)d_in[6];  // (F)
    const float* w2        = (const float*)d_in[7];  // (F,F)
    const float* b2        = (const float*)d_in[8];  // (F)
    float* out             = (float*)d_out;          // (N,F,T)

    const long nFT = (long)N * F * T;   // 16,777,216
    const long nTC = (long)N * T * C;   // 16,777,216
    const long nFF = (long)N * F * F;   //  4,194,304

    char* w = (char*)d_ws;
    unsigned short* ft_bf   = (unsigned short*)w; w += nFT * 2;      // (N,F,T)
    unsigned short* fsrc_bf = (unsigned short*)w; w += nTC * 2;      // (N,T,C)
    unsigned short* fs_bf   = (unsigned short*)w; w += nFT * 2;      // (N,F,T)
    float*          scores  = (float*)w;          w += nFF * 4;      // (N,F,F)
    unsigned short* attn_bf = (unsigned short*)w; w += nFF * 2;      // (N,F,F)
    unsigned short* x_bf    = (unsigned short*)w; w += nFT * 2;      // (N,T,F)
    unsigned short* h_bf    = (unsigned short*)w; w += nFT * 2;      // (N,T,F)
    unsigned short* cw_bf   = (unsigned short*)w; w += (long)F * C * 2;
    unsigned short* w1_bf   = (unsigned short*)w; w += (long)F * F * 2;
    unsigned short* w2_bf   = (unsigned short*)w; w += (long)F * F * 2;

    dim3 blk(256);

    // Pre-convert all GEMM operands to bf16 (float4 / ushort4 vectorized).
    f32_to_bf16_v4_kernel<<<dim3((unsigned)(nFT / 4 / 256)), blk, 0, stream>>>(
        (const v4f*)feature_t, (us4*)ft_bf, nFT / 4);
    f32_to_bf16_v4_kernel<<<dim3((unsigned)(nTC / 4 / 256)), blk, 0, stream>>>(
        (const v4f*)feature_s, (us4*)fsrc_bf, nTC / 4);
    f32_to_bf16_v4_kernel<<<dim3((F * C / 4 + 255) / 256), blk, 0, stream>>>(
        (const v4f*)conv_w, (us4*)cw_bf, (long)F * C / 4);
    f32_to_bf16_v4_kernel<<<dim3((F * F / 4 + 255) / 256), blk, 0, stream>>>(
        (const v4f*)w1, (us4*)w1_bf, (long)F * F / 4);
    f32_to_bf16_v4_kernel<<<dim3((F * F / 4 + 255) / 256), blk, 0, stream>>>(
        (const v4f*)w2, (us4*)w2_bf, (long)F * F / 4);

    // GEMM1: fs(F,T) = conv_w x feature_s^T + conv_b   [B^T = fsrc (T,C)]
    gemm_bf16_wmma_kernel<0, 0><<<dim3(T / 128, F / 128, N), blk, 0, stream>>>(
        cw_bf, fsrc_bf, fs_bf, conv_b, nullptr, nullptr, nullptr,
        1.0f, C, C, C, T, 0, 0L, (long)T * C, (long)F * T, 0L, T);

    // GEMM2: scores(F,F) = fs x ft^T * (1/T)           [B^T = ft (F,T)]
    gemm_bf16_wmma_kernel<0, 1><<<dim3(F / 128, F / 128, N), blk, 0, stream>>>(
        fs_bf, ft_bf, scores, nullptr, nullptr, nullptr, nullptr,
        1.0f / (float)T, T, T, T, F, 0, (long)F * T, (long)F * T, (long)F * F, 0L, T);

    // softmax over last dim of scores -> attn (bf16)
    softmax512_kernel<<<dim3(N * F), blk, 0, stream>>>(scores, attn_bf);

    // GEMM3: cross(F,T) = attn x ft + ft; stored transposed as x(T,F)
    gemm_bf16_wmma_kernel<1, 2><<<dim3(T / 128, F / 128, N), blk, 0, stream>>>(
        attn_bf, ft_bf, x_bf, nullptr, nullptr, feature_t, nullptr,
        1.0f, F, F, T, F, T, (long)F * F, (long)F * T, (long)T * F, (long)F * T, T);

    // GEMM4: h(T,F) = gelu(x x w1^T + b1)              [B^T = w1 (F,F)]
    gemm_bf16_wmma_kernel<0, 3><<<dim3(F / 128, T / 128, N), blk, 0, stream>>>(
        x_bf, w1_bf, h_bf, nullptr, b1, nullptr, nullptr,
        1.0f, F, F, F, F, 0, (long)T * F, 0L, (long)T * F, 0L, T);

    // GEMM5: y(F,T) = w2 x h^T + b2 + ft, * mask       [B^T = h (T,F)]
    gemm_bf16_wmma_kernel<0, 4><<<dim3(T / 128, F / 128, N), blk, 0, stream>>>(
        w2_bf, h_bf, out, b2, nullptr, feature_t, maskp,
        1.0f, F, F, F, T, T, 0L, (long)T * F, (long)F * T, (long)F * T, T);
}
